// NodeModel_64854006170307
// MI455X (gfx1250) — matrix operand order repaired and checked by reading, not compiled, not run on hardware
//
#include <hip/hip_runtime.h>

typedef __attribute__((ext_vector_type(16))) _Float16 v16h;
typedef __attribute__((ext_vector_type(8)))  float    v8f;

#define D_NODE  64
#define D_EDGE  32
#define D_GLOB  32
#define OUT_DIM 64
#define IN1     192   // 6 K-chunks of 32
#define IN2     160   // 5 K-chunks of 32
#define WSTRIDE 200   // padded halves per W^T row in LDS: 400B rows -> 16B-aligned frags,
                      // lane bank stride 36 dw -> conflict-free b128 fragment reads

union Frag16 { uint4 q[2]; v16h h; };

// ---- B fragment: 32x16 f16 tile of W. Lane n%16 = column; elements 0..7 hold
// K=kb..kb+7, elements 8..15 hold K=kb+16..kb+23 (caller folds +8*lh into kbase).
// W^T stored [64][WSTRIDE] halves in LDS -> two runs of 8 contiguous halves per lane.
__device__ inline v16h load_b_frag(const _Float16* __restrict__ lds_w, int col, int kbase) {
    Frag16 f;
    const uint4* p = (const uint4*)(lds_w + col * WSTRIDE + kbase);
    f.q[0] = p[0];
    f.q[1] = p[2];
    return f.h;
}

// ---- A fragment (16-bit A 16x32 layout): elements 0..7 = K+8*lh .. +7,
// elements 8..15 = K+16+8*lh .. +7  (p points at this row's 32-wide chunk).
__device__ inline v16h load_a_frag_h(const _Float16* __restrict__ p, int lh) {
    Frag16 f;
    const uint4* q = (const uint4*)(p + lh * 8);
    f.q[0] = q[0];          // halves ka .. ka+7
    f.q[1] = q[2];          // halves ka+16 .. ka+23
    return f.h;
}

__device__ inline v16h load_a_frag_f32(const float* __restrict__ p, int lh) {
    const int ka = lh * 8;
    float4 a0 = *(const float4*)(p + ka);
    float4 a1 = *(const float4*)(p + ka + 4);
    float4 b0 = *(const float4*)(p + ka + 16);
    float4 b1 = *(const float4*)(p + ka + 20);
    v16h h;
    h[0]  = (_Float16)a0.x; h[1]  = (_Float16)a0.y; h[2]  = (_Float16)a0.z; h[3]  = (_Float16)a0.w;
    h[4]  = (_Float16)a1.x; h[5]  = (_Float16)a1.y; h[6]  = (_Float16)a1.z; h[7]  = (_Float16)a1.w;
    h[8]  = (_Float16)b0.x; h[9]  = (_Float16)b0.y; h[10] = (_Float16)b0.z; h[11] = (_Float16)b0.w;
    h[12] = (_Float16)b1.x; h[13] = (_Float16)b1.y; h[14] = (_Float16)b1.z; h[15] = (_Float16)b1.w;
    return h;
}

__global__ void zero_f4_kernel(float4* __restrict__ p, int n4) {
    int i = blockIdx.x * 256 + threadIdx.x;
    if (i < n4) p[i] = make_float4(0.f, 0.f, 0.f, 0.f);
}

// f32 -> f16 streaming convert, 8 elements/thread (cvt_pk + b128 store)
union Pack8 { uint4 q; _Float16 h[8]; };
__global__ void cvt_f16_kernel(const float4* __restrict__ in, uint4* __restrict__ out, int n8) {
    int i = blockIdx.x * 256 + threadIdx.x;
    if (i >= n8) return;
    float4 a = in[2 * i], b = in[2 * i + 1];
    Pack8 p;
    p.h[0] = (_Float16)a.x; p.h[1] = (_Float16)a.y; p.h[2] = (_Float16)a.z; p.h[3] = (_Float16)a.w;
    p.h[4] = (_Float16)b.x; p.h[5] = (_Float16)b.y; p.h[6] = (_Float16)b.z; p.h[7] = (_Float16)b.w;
    out[i] = p.q;
}

// One K-chunk step: batch-load all 4 B fragments (8 ds_load_b128 in one clause,
// single dscnt wait), then issue 4 independent WMMAs.
__device__ inline void wmma_kchunk(const _Float16* __restrict__ lds_w, int lr, int kb,
                                   const v16h& a, v8f acc[4]) {
    v16h b0 = load_b_frag(lds_w, 0 * 16 + lr, kb);
    v16h b1 = load_b_frag(lds_w, 1 * 16 + lr, kb);
    v16h b2 = load_b_frag(lds_w, 2 * 16 + lr, kb);
    v16h b3 = load_b_frag(lds_w, 3 * 16 + lr, kb);
    acc[0] = __builtin_amdgcn_wmma_f32_16x16x32_f16(false, a, false, b0, (short)0, acc[0], false, false);
    acc[1] = __builtin_amdgcn_wmma_f32_16x16x32_f16(false, a, false, b1, (short)0, acc[1], false, false);
    acc[2] = __builtin_amdgcn_wmma_f32_16x16x32_f16(false, a, false, b2, (short)0, acc[2], false, false);
    acc[3] = __builtin_amdgcn_wmma_f32_16x16x32_f16(false, a, false, b3, (short)0, acc[3], false, false);
}

// ---------------- Edge MLP + scatter-sum -------------------------------------
// One wave = 16 edges: A = [16 x 192] gathered concat (f16), B = W1 [192 x 64],
// f32 acc, bias+ReLU, global_atomic_add_f32 scatter into agg[dst].
template <bool PRECONV>
__global__ __launch_bounds__(256) void edge_mlp_scatter(
    const float* __restrict__ x, const _Float16* __restrict__ xh,
    const int* __restrict__ ei, const float* __restrict__ ea,
    const float* __restrict__ u, const _Float16* __restrict__ uh,
    const int* __restrict__ batch, const float* __restrict__ W1,
    const float* __restrict__ b1, float* __restrict__ agg, int n_edges) {

    __shared__ _Float16 lds_w[OUT_DIM * WSTRIDE];
    __shared__ float    lds_b[OUT_DIM];

    const int tid = threadIdx.x;
    for (int i = tid; i < IN1 * OUT_DIM; i += 256) {   // W1 [192][64] -> W1^T in LDS
        int k = i >> 6, n = i & 63;
        lds_w[n * WSTRIDE + k] = (_Float16)W1[i];
    }
    if (tid < OUT_DIM) lds_b[tid] = b1[tid];
    __syncthreads();

    const int wave = tid >> 5, lane = tid & 31;
    const int lh = lane >> 4, lr = lane & 15;
    const int tile = blockIdx.x * 8 + wave;
    const int ntiles = n_edges >> 4;
    if (tile >= ntiles) return;                         // wave-uniform: EXEC stays all-1s

    const int e   = tile * 16 + lr;                     // this lane's A-row
    const int src = ei[e];
    const int dst = ei[n_edges + e];
    const int bg  = batch[dst];

    // Build the six 16x32 A fragments of the gathered concat row
    v16h a[6];
    if (PRECONV) {
        const _Float16* xd = xh + (size_t)dst * D_NODE;
        const _Float16* xs = xh + (size_t)src * D_NODE;
        a[0] = load_a_frag_h(xd,      lh);              // cols   0..31  x[dst]
        a[1] = load_a_frag_h(xd + 32, lh);              // cols  32..63
        a[2] = load_a_frag_h(xs,      lh);              // cols  64..95  x[src]
        a[3] = load_a_frag_h(xs + 32, lh);              // cols  96..127
        a[5] = load_a_frag_h(uh + (size_t)bg * D_GLOB, lh); // cols 160..191 u[batch[dst]]
    } else {
        const float* xd = x + (size_t)dst * D_NODE;
        const float* xs = x + (size_t)src * D_NODE;
        a[0] = load_a_frag_f32(xd,      lh);
        a[1] = load_a_frag_f32(xd + 32, lh);
        a[2] = load_a_frag_f32(xs,      lh);
        a[3] = load_a_frag_f32(xs + 32, lh);
        a[5] = load_a_frag_f32(u + (size_t)bg * D_GLOB, lh);
    }
    a[4] = load_a_frag_f32(ea + (size_t)e * D_EDGE, lh);    // cols 128..159 (streamed once)

    v8f acc[4] = {};
#pragma unroll
    for (int kc = 0; kc < 6; ++kc)
        wmma_kchunk(lds_w, lr, kc * 32 + lh * 8, a[kc], acc);

    // destination nodes for rows m = lh*8 + 0..7 of this tile
    int drow[8];
#pragma unroll
    for (int j = 0; j < 8; ++j)
        drow[j] = ei[n_edges + tile * 16 + lh * 8 + j];

#pragma unroll
    for (int nc = 0; nc < 4; ++nc) {
        const int n = nc * 16 + lr;
        const float bias = lds_b[n];
#pragma unroll
        for (int j = 0; j < 8; ++j) {
            float v = acc[nc][j] + bias;
            v = v > 0.f ? v : 0.f;
            unsafeAtomicAdd(agg + (size_t)drow[j] * OUT_DIM + n, v);  // global_atomic_add_f32
        }
    }
}

// ---------------- Node MLP ---------------------------------------------------
// One wave = 16 nodes: A = [16 x 160] concat [x, agg, u[batch]], B = W2 [160 x 64].
template <bool PRECONV>
__global__ __launch_bounds__(256) void node_mlp(
    const float* __restrict__ x, const _Float16* __restrict__ xh,
    const float* __restrict__ u, const _Float16* __restrict__ uh,
    const int* __restrict__ batch, const float* __restrict__ agg,
    const float* __restrict__ W2, const float* __restrict__ b2,
    float* __restrict__ out, int n_nodes) {

    __shared__ _Float16 lds_w[OUT_DIM * WSTRIDE];
    __shared__ float    lds_b[OUT_DIM];

    const int tid = threadIdx.x;
    for (int i = tid; i < IN2 * OUT_DIM; i += 256) {
        int k = i >> 6, n = i & 63;
        lds_w[n * WSTRIDE + k] = (_Float16)W2[i];
    }
    if (tid < OUT_DIM) lds_b[tid] = b2[tid];
    __syncthreads();

    const int wave = tid >> 5, lane = tid & 31;
    const int lh = lane >> 4, lr = lane & 15;
    const int tile = blockIdx.x * 8 + wave;
    const int ntiles = n_nodes >> 4;
    if (tile >= ntiles) return;

    const int node = tile * 16 + lr;
    const int bg = batch[node];

    v16h a[5];
    if (PRECONV) {
        const _Float16* xn = xh + (size_t)node * D_NODE;
        a[0] = load_a_frag_h(xn,      lh);              // cols   0..31  x
        a[1] = load_a_frag_h(xn + 32, lh);              // cols  32..63
        a[4] = load_a_frag_h(uh + (size_t)bg * D_GLOB, lh); // cols 128..159 u
    } else {
        const float* xn = x + (size_t)node * D_NODE;
        a[0] = load_a_frag_f32(xn,      lh);
        a[1] = load_a_frag_f32(xn + 32, lh);
        a[4] = load_a_frag_f32(u + (size_t)bg * D_GLOB, lh);
    }
    const float* an = agg + (size_t)node * OUT_DIM;     // agg read once: convert inline
    a[2] = load_a_frag_f32(an,      lh);                // cols  64..95
    a[3] = load_a_frag_f32(an + 32, lh);                // cols  96..127

    v8f acc[4] = {};
#pragma unroll
    for (int kc = 0; kc < 5; ++kc)
        wmma_kchunk(lds_w, lr, kc * 32 + lh * 8, a[kc], acc);

#pragma unroll
    for (int nc = 0; nc < 4; ++nc) {
        const int n = nc * 16 + lr;
        const float bias = lds_b[n];
#pragma unroll
        for (int j = 0; j < 8; ++j) {
            const int m = lh * 8 + j;
            float v = acc[nc][j] + bias;
            v = v > 0.f ? v : 0.f;
            out[(size_t)(tile * 16 + m) * OUT_DIM + n] = v;
        }
    }
}

static inline size_t align256(size_t v) { return (v + 255) & ~(size_t)255; }

extern "C" void kernel_launch(void* const* d_in, const int* in_sizes, int n_in,
                              void* d_out, int out_size, void* d_ws, size_t ws_size,
                              hipStream_t stream) {
    const float* x     = (const float*)d_in[0];
    const int*   ei    = (const int*)  d_in[1];
    const float* ea    = (const float*)d_in[2];
    const float* u     = (const float*)d_in[3];
    const int*   batch = (const int*)  d_in[4];
    const float* W1    = (const float*)d_in[5];
    const float* b1    = (const float*)d_in[6];
    const float* W2    = (const float*)d_in[7];
    const float* b2    = (const float*)d_in[8];

    const int N = in_sizes[0] / D_NODE;     // 100000
    const int E = in_sizes[2] / D_EDGE;     // 1000000
    const int G = in_sizes[3] / D_GLOB;     // 16
    float* out = (float*)d_out;

    const size_t agg_bytes = (size_t)N * OUT_DIM * sizeof(float);
    const size_t xh_bytes  = (size_t)N * D_NODE * sizeof(_Float16);
    const size_t uh_bytes  = (size_t)G * D_GLOB * sizeof(_Float16);
    const size_t xh_off    = align256(agg_bytes);
    const size_t uh_off    = align256(xh_off + xh_bytes);
    const bool   preconv   = ws_size >= uh_off + uh_bytes;

    float*     agg = (ws_size >= agg_bytes) ? (float*)d_ws : out;  // in-place fallback safe
    _Float16*  xh  = preconv ? (_Float16*)((char*)d_ws + xh_off) : nullptr;
    _Float16*  uh  = preconv ? (_Float16*)((char*)d_ws + uh_off) : nullptr;

    const int n4 = (N * OUT_DIM) / 4;
    zero_f4_kernel<<<(n4 + 255) / 256, 256, 0, stream>>>((float4*)agg, n4);

    if (preconv) {
        const int nx8 = (N * D_NODE) / 8;
        cvt_f16_kernel<<<(nx8 + 255) / 256, 256, 0, stream>>>(
            (const float4*)x, (uint4*)xh, nx8);
        const int nu8 = (G * D_GLOB) / 8;
        cvt_f16_kernel<<<(nu8 + 255) / 256, 256, 0, stream>>>(
            (const float4*)u, (uint4*)uh, nu8);
    }

    const int eb = (E / 16 + 7) / 8;
    const int nb = (N / 16 + 7) / 8;
    if (preconv) {
        edge_mlp_scatter<true><<<eb, 256, 0, stream>>>(x, xh, ei, ea, u, uh, batch, W1, b1, agg, E);
        node_mlp<true><<<nb, 256, 0, stream>>>(x, xh, u, uh, batch, agg, W2, b2, out, N);
    } else {
        edge_mlp_scatter<false><<<eb, 256, 0, stream>>>(x, xh, ei, ea, u, uh, batch, W1, b1, agg, E);
        node_mlp<false><<<nb, 256, 0, stream>>>(x, xh, u, uh, batch, agg, W2, b2, out, N);
    }
}
